// Model_59665685676339
// MI455X (gfx1250) — matrix operand order repaired and checked by reading, not compiled, not run on hardware
//
#include <hip/hip_runtime.h>
#include <stdint.h>

// ---------------- constants ----------------
#define EMBD   300
#define E2D    600
#define NLAYER 5
#define NNODES 100000
#define NEDGES 400000
#define NGRAPH 1024
#define KP1    320          // 300 padded to mult of 32
#define KP2    608          // 600 padded to mult of 32
#define MPAD   100032       // 100000 padded to mult of 64
#define NPB1   640          // 600 padded to mult of 128 (Bt rows)
#define NPB2   384          // 300 padded to mult of 128 (Bt rows)
#define BN_EPS_F 1e-5f
#define INV_TEMP 25.0f

typedef unsigned short u16;
typedef __attribute__((ext_vector_type(8)))  __bf16 v8bf;
typedef __attribute__((ext_vector_type(16))) __bf16 v16bf;
typedef __attribute__((ext_vector_type(8)))  float  v8f;

// ---------------- helpers ----------------
__device__ __forceinline__ u16 f2bf(float x) {
    unsigned u = __float_as_uint(x);
    unsigned r = u + 0x7FFFu + ((u >> 16) & 1u);   // round-to-nearest-even
    return (u16)(r >> 16);
}

__device__ __forceinline__ v16bf cat8(v8bf lo, v8bf hi) {
    return __builtin_shufflevector(lo, hi, 0,1,2,3,4,5,6,7,8,9,10,11,12,13,14,15);
}

// ---------------- zero kernels ----------------
__global__ void gnn_zero16(uint4* p, size_t n16) {
    size_t i = (size_t)blockIdx.x * blockDim.x + threadIdx.x;
    size_t stride = (size_t)gridDim.x * blockDim.x;
    uint4 z; z.x = z.y = z.z = z.w = 0u;
    for (; i < n16; i += stride) p[i] = z;
}

__global__ void gnn_zero_f32(float* p, int n) {
    int i = blockIdx.x * blockDim.x + threadIdx.x;
    if (i < n) p[i] = 0.0f;
}

// ---------------- weight transpose: W[K,N] f32 -> Bt[n*Kp + k] bf16 ----------------
__global__ void gnn_transpose_w(const float* __restrict__ W, u16* __restrict__ Bt,
                                int K, int N, int Kp) {
    int idx = blockIdx.x * blockDim.x + threadIdx.x;
    if (idx >= N * K) return;
    int n = idx / K, k = idx % K;
    Bt[(size_t)n * Kp + k] = f2bf(W[(size_t)k * N + n]);
}

// ---------------- node embedding init ----------------
__global__ void gnn_embed(const int* __restrict__ x, const float* __restrict__ ae1,
                          const float* __restrict__ ae2, float* __restrict__ h) {
    int i = blockIdx.x * blockDim.x + threadIdx.x;
    if (i >= NNODES * EMBD) return;
    int n = i / EMBD, f = i - n * EMBD;
    h[i] = ae1[(size_t)x[2 * n] * EMBD + f] + ae2[(size_t)x[2 * n + 1] * EMBD + f];
}

// ---------------- agg = h + self_emb ----------------
__global__ void gnn_agg_init(const float* __restrict__ h, const float* __restrict__ be1,
                             const float* __restrict__ be2, float* __restrict__ agg) {
    int i = blockIdx.x * blockDim.x + threadIdx.x;
    if (i >= NNODES * EMBD) return;
    int f = i % EMBD;
    agg[i] = h[i] + be1[4 * EMBD + f] + be2[f];   // self-loop bond type 4 / 0
}

// ---------------- edge scatter: agg[dst] += h[src] + e_emb  (one wave per edge) ----------------
__global__ void gnn_edge_scatter(const int* __restrict__ ei, const int* __restrict__ ea,
                                 const float* __restrict__ h, const float* __restrict__ be1,
                                 const float* __restrict__ be2, float* __restrict__ agg) {
    int e = blockIdx.x * (blockDim.x >> 5) + (threadIdx.x >> 5);
    if (e >= NEDGES) return;
    int lane = threadIdx.x & 31;
    int src = ei[e], dst = ei[NEDGES + e];
    const float* hs = h + (size_t)src * EMBD;
    const float* e1 = be1 + (size_t)ea[2 * e] * EMBD;
    const float* e2 = be2 + (size_t)ea[2 * e + 1] * EMBD;
    float* ad = agg + (size_t)dst * EMBD;
    for (int f = lane; f < EMBD; f += 32)
        atomicAdd(ad + f, hs[f] + e1[f] + e2[f]);
}

// ---------------- f32 [M,K] -> bf16 padded [M,Kp] ----------------
__global__ void gnn_to_bf16(const float* __restrict__ src, u16* __restrict__ dst,
                            int M, int K, int Kp) {
    int i = blockIdx.x * blockDim.x + threadIdx.x;
    if (i >= M * K) return;
    int r = i / K, c = i - r * K;
    dst[(size_t)r * Kp + c] = f2bf(src[i]);
}

// ---------------- WMMA GEMM: Out[M,N] = act(A[M,Kp] @ Bt[N,Kp]^T + bias) * scale ----------------
// A, Bt are bf16; block tile 64x128, wave tile 32x32 (4 WMMAs per K-step).
template <bool OUT_BF16, bool RELU, bool HAS_BIAS>
__global__ __launch_bounds__(256)
void gnn_wmma_gemm(const u16* __restrict__ A, const u16* __restrict__ Bt,
                   const float* __restrict__ bias, void* __restrict__ Out,
                   int M, int N, int Kp, int ldo, float scale) {
    const int lane = threadIdx.x & 31;
    const int wave = threadIdx.x >> 5;
    const int l15  = lane & 15;
    const int hf   = lane >> 4;                 // lane half (0/1)
    const int m0 = blockIdx.y * 64  + (wave >> 2) * 32;
    const int n0 = blockIdx.x * 128 + (wave & 3) * 32;

    // A 16x32 bf16 layout: lane half selects K-chunk {0,8}; chunks at K and K+16.
    const u16* a0p = A  + (size_t)(m0 + l15)      * Kp + hf * 8;
    const u16* a1p = A  + (size_t)(m0 + 16 + l15) * Kp + hf * 8;
    // B 32x16 bf16 layout: lane half selects K-chunk {0,8}; chunks at K and K+16.
    const u16* b0p = Bt + (size_t)(n0 + l15)      * Kp + hf * 8;
    const u16* b1p = Bt + (size_t)(n0 + 16 + l15) * Kp + hf * 8;

    v8f c00 = {}, c01 = {}, c10 = {}, c11 = {};
    for (int k = 0; k < Kp; k += 32) {
        __builtin_prefetch(a0p + k + 256, 0, 0);
        __builtin_prefetch(b0p + k + 256, 0, 0);
        v16bf a0 = cat8(*(const v8bf*)(a0p + k), *(const v8bf*)(a0p + k + 16));
        v16bf a1 = cat8(*(const v8bf*)(a1p + k), *(const v8bf*)(a1p + k + 16));
        v16bf b0 = cat8(*(const v8bf*)(b0p + k), *(const v8bf*)(b0p + k + 16));
        v16bf b1 = cat8(*(const v8bf*)(b1p + k), *(const v8bf*)(b1p + k + 16));
        c00 = __builtin_amdgcn_wmma_f32_16x16x32_bf16(false, a0, false, b0, (short)0, c00, false, false);
        c01 = __builtin_amdgcn_wmma_f32_16x16x32_bf16(false, a0, false, b1, (short)0, c01, false, false);
        c10 = __builtin_amdgcn_wmma_f32_16x16x32_bf16(false, a1, false, b0, (short)0, c10, false, false);
        c11 = __builtin_amdgcn_wmma_f32_16x16x32_bf16(false, a1, false, b1, (short)0, c11, false, false);
    }

    auto store_tile = [&](v8f c, int mt, int nt) {
        int col = nt + l15;
        if (col >= N) return;
        float b = HAS_BIAS ? bias[col] : 0.0f;
        int rowBase = mt + hf * 8;
        #pragma unroll
        for (int r = 0; r < 8; ++r) {
            int row = rowBase + r;
            if (row >= M) break;
            float v = (c[r] + b) * scale;
            if (RELU) v = fmaxf(v, 0.0f);
            if (OUT_BF16) ((u16*)Out)[(size_t)row * ldo + col] = f2bf(v);
            else          ((float*)Out)[(size_t)row * ldo + col] = v;
        }
    };
    store_tile(c00, m0,      n0);
    store_tile(c01, m0,      n0 + 16);
    store_tile(c10, m0 + 16, n0);
    store_tile(c11, m0 + 16, n0 + 16);
}

// ---------------- BatchNorm ----------------
__global__ void gnn_bn_stats(const float* __restrict__ hr, float* __restrict__ stats,
                             int rows_per_block) {
    int f = threadIdx.x;
    if (f >= EMBD) return;
    int r0 = blockIdx.x * rows_per_block;
    int r1 = min(r0 + rows_per_block, NNODES);
    float s = 0.f, ss = 0.f;
    for (int r = r0; r < r1; ++r) {
        float v = hr[(size_t)r * EMBD + f];
        s += v; ss += v * v;
    }
    atomicAdd(&stats[f], s);
    atomicAdd(&stats[EMBD + f], ss);
}

__global__ void gnn_bn_coef(const float* __restrict__ stats, const float* __restrict__ scale,
                            const float* __restrict__ bias, float* __restrict__ coef) {
    int f = threadIdx.x;
    if (f >= EMBD) return;
    float mean = stats[f] * (1.0f / NNODES);
    float var  = stats[EMBD + f] * (1.0f / NNODES) - mean * mean;
    float a = rsqrtf(var + BN_EPS_F) * scale[f];
    coef[f] = a;
    coef[EMBD + f] = bias[f] - mean * a;
}

__global__ void gnn_bn_apply(const float* __restrict__ hr, const float* __restrict__ coef,
                             float* __restrict__ h, int do_relu) {
    int i = blockIdx.x * blockDim.x + threadIdx.x;
    if (i >= NNODES * EMBD) return;
    int f = i % EMBD;
    float v = hr[i] * coef[f] + coef[EMBD + f];
    if (do_relu) v = fmaxf(v, 0.0f);
    h[i] = v;
}

// ---------------- pooling ----------------
__global__ void gnn_pool_count(const int* __restrict__ batch, float* __restrict__ counts) {
    int n = blockIdx.x * blockDim.x + threadIdx.x;
    if (n < NNODES) atomicAdd(&counts[batch[n]], 1.0f);
}

__global__ void gnn_pool_sum(const float* __restrict__ h, const int* __restrict__ batch,
                             float* __restrict__ pooled) {
    int i = blockIdx.x * blockDim.x + threadIdx.x;
    if (i >= NNODES * EMBD) return;
    int n = i / EMBD, f = i - n * EMBD;
    atomicAdd(&pooled[(size_t)batch[n] * EMBD + f], h[i]);
}

__global__ void gnn_pool_div_bf16(const float* __restrict__ pooled, const float* __restrict__ counts,
                                  u16* __restrict__ dst) {
    int i = blockIdx.x * blockDim.x + threadIdx.x;
    if (i >= NGRAPH * EMBD) return;
    int g = i / EMBD, f = i - g * EMBD;
    dst[(size_t)g * KP1 + f] = f2bf(pooled[i] / fmaxf(counts[g], 1.0f));
}

// ---------------- L2 row normalize proj -> bf16 feats (one wave per row) ----------------
__global__ void gnn_normalize(const float* __restrict__ proj, u16* __restrict__ featsbf) {
    int row = blockIdx.x;
    int lane = threadIdx.x;
    const float* p = proj + (size_t)row * EMBD;
    float ss = 0.f;
    for (int f = lane; f < EMBD; f += 32) { float v = p[f]; ss += v * v; }
    for (int m = 16; m >= 1; m >>= 1) ss += __shfl_xor(ss, m, 32);
    float inv = 1.0f / fmaxf(sqrtf(ss), 1e-12f);
    for (int f = lane; f < EMBD; f += 32)
        featsbf[(size_t)row * KP1 + f] = f2bf(p[f] * inv);
}

// ---------------- host ----------------
extern "C" void kernel_launch(void* const* d_in, const int* in_sizes, int n_in,
                              void* d_out, int out_size, void* d_ws, size_t ws_size,
                              hipStream_t stream) {
    const int*   x    = (const int*)d_in[0];
    const int*   ei   = (const int*)d_in[1];
    const int*   ea   = (const int*)d_in[2];
    const int*   batch= (const int*)d_in[3];
    const float* ae1  = (const float*)d_in[4];
    const float* ae2  = (const float*)d_in[5];
    const float* be1  = (const float*)d_in[6];   // [5,6,300]
    const float* be2  = (const float*)d_in[7];   // [5,3,300]
    const float* W1   = (const float*)d_in[8];   // [5,300,600]
    const float* b1   = (const float*)d_in[9];   // [5,600]
    const float* W2   = (const float*)d_in[10];  // [5,600,300]
    const float* b2   = (const float*)d_in[11];  // [5,300]
    const float* bnS  = (const float*)d_in[12];
    const float* bnB  = (const float*)d_in[13];
    const float* pW1  = (const float*)d_in[14];
    const float* pb1  = (const float*)d_in[15];
    const float* pW2  = (const float*)d_in[16];
    const float* pb2  = (const float*)d_in[17];

    uint8_t* ws = (uint8_t*)d_ws;
    size_t off = 0;
    auto carve = [&](size_t bytes) -> uint8_t* {
        uint8_t* p = ws + off;
        off += (bytes + 255) & ~(size_t)255;
        return p;
    };
    float* h      = (float*)carve((size_t)NNODES * EMBD * 4);
    float* agg    = (float*)carve((size_t)NNODES * EMBD * 4);      // also reused as h_raw
    u16*   Abf    = (u16*)  carve((size_t)MPAD * KP1 * 2);
    u16*   Ybf    = (u16*)  carve((size_t)MPAD * KP2 * 2);
    u16*   W1t    = (u16*)  carve((size_t)NLAYER * NPB1 * KP1 * 2);
    u16*   W2t    = (u16*)  carve((size_t)NLAYER * NPB2 * KP2 * 2);
    u16*   pW1t   = (u16*)  carve((size_t)NPB2 * KP1 * 2);
    u16*   pW2t   = (u16*)  carve((size_t)NPB2 * KP1 * 2);
    float* stats  = (float*)carve(2 * EMBD * 4);
    float* coef   = (float*)carve(2 * EMBD * 4);
    float* counts = (float*)carve(NGRAPH * 4);
    float* pooled = (float*)carve((size_t)NGRAPH * EMBD * 4);
    u16*   pooledbf=(u16*)  carve((size_t)NGRAPH * KP1 * 2);
    u16*   P1bf   = (u16*)  carve((size_t)NGRAPH * KP1 * 2);
    float* proj   = (float*)carve((size_t)NGRAPH * EMBD * 4);
    u16*   featsbf= (u16*)  carve((size_t)NGRAPH * KP1 * 2);
    size_t total = off;

    const int NE = NNODES * EMBD;                // 30,000,000
    dim3 blk(256);
    auto nb = [](long long n, int t) { return (unsigned)((n + t - 1) / t); };

    // zero all workspace (padding regions must be zero for unguarded WMMA loads)
    gnn_zero16<<<4096, 256, 0, stream>>>((uint4*)ws, total / 16);

    // weight transposes to bf16 [n,k]
    for (int l = 0; l < NLAYER; ++l) {
        gnn_transpose_w<<<nb(E2D * EMBD, 256), blk, 0, stream>>>(
            W1 + (size_t)l * EMBD * E2D, W1t + (size_t)l * NPB1 * KP1, EMBD, E2D, KP1);
        gnn_transpose_w<<<nb(EMBD * E2D, 256), blk, 0, stream>>>(
            W2 + (size_t)l * E2D * EMBD, W2t + (size_t)l * NPB2 * KP2, E2D, EMBD, KP2);
    }
    gnn_transpose_w<<<nb(EMBD * EMBD, 256), blk, 0, stream>>>(pW1, pW1t, EMBD, EMBD, KP1);
    gnn_transpose_w<<<nb(EMBD * EMBD, 256), blk, 0, stream>>>(pW2, pW2t, EMBD, EMBD, KP1);

    // initial node embeddings
    gnn_embed<<<nb(NE, 256), blk, 0, stream>>>(x, ae1, ae2, h);

    for (int l = 0; l < NLAYER; ++l) {
        const float* be1l = be1 + (size_t)l * 6 * EMBD;
        const float* be2l = be2 + (size_t)l * 3 * EMBD;

        gnn_agg_init<<<nb(NE, 256), blk, 0, stream>>>(h, be1l, be2l, agg);
        gnn_edge_scatter<<<NEDGES / 8, blk, 0, stream>>>(ei, ea, h, be1l, be2l, agg);

        gnn_to_bf16<<<nb(NE, 256), blk, 0, stream>>>(agg, Abf, NNODES, EMBD, KP1);

        // Y = relu(agg @ W1 + b1) -> bf16 [MPAD, 608]
        dim3 g1((E2D + 127) / 128, MPAD / 64);
        gnn_wmma_gemm<true, true, true><<<g1, blk, 0, stream>>>(
            Abf, W1t + (size_t)l * NPB1 * KP1, b1 + (size_t)l * E2D,
            Ybf, NNODES, E2D, KP1, KP2, 1.0f);

        // h_raw = Y @ W2 + b2 -> f32 [100000, 300] (reuse agg)
        dim3 g2((EMBD + 127) / 128, MPAD / 64);
        gnn_wmma_gemm<false, false, true><<<g2, blk, 0, stream>>>(
            Ybf, W2t + (size_t)l * NPB2 * KP2, b2 + (size_t)l * EMBD,
            agg, NNODES, EMBD, KP2, EMBD, 1.0f);

        // batch norm
        gnn_zero_f32<<<3, 256, 0, stream>>>(stats, 2 * EMBD);
        gnn_bn_stats<<<250, 320, 0, stream>>>(agg, stats, 400);
        gnn_bn_coef<<<1, 320, 0, stream>>>(stats, bnS + (size_t)l * EMBD, bnB + (size_t)l * EMBD, coef);
        gnn_bn_apply<<<nb(NE, 256), blk, 0, stream>>>(agg, coef, h, (l < NLAYER - 1) ? 1 : 0);
    }

    // mean pooling per graph
    gnn_pool_count<<<nb(NNODES, 256), blk, 0, stream>>>(batch, counts);
    gnn_pool_sum<<<nb(NE, 256), blk, 0, stream>>>(h, batch, pooled);
    gnn_pool_div_bf16<<<nb(NGRAPH * EMBD, 256), blk, 0, stream>>>(pooled, counts, pooledbf);

    // projection head
    dim3 gp((EMBD + 127) / 128, NGRAPH / 64);
    gnn_wmma_gemm<true, true, true><<<gp, blk, 0, stream>>>(
        pooledbf, pW1t, pb1, P1bf, NGRAPH, EMBD, KP1, KP1, 1.0f);
    gnn_wmma_gemm<false, false, true><<<gp, blk, 0, stream>>>(
        P1bf, pW2t, pb2, proj, NGRAPH, EMBD, KP1, EMBD, 1.0f);

    // L2 normalize rows -> bf16 feats
    gnn_normalize<<<NGRAPH, 32, 0, stream>>>(proj, featsbf);

    // logits = f0 @ f1^T / TEMP  -> d_out [512, 512] f32
    dim3 gl((512 + 127) / 128, 512 / 64);
    gnn_wmma_gemm<false, false, false><<<gl, blk, 0, stream>>>(
        featsbf, featsbf + (size_t)512 * KP1, nullptr,
        (float*)d_out, 512, 512, KP1, 512, INV_TEMP);

    (void)in_sizes; (void)n_in; (void)out_size; (void)ws_size;
}
